// Discriminator_429496729631
// MI455X (gfx1250) — matrix operand order repaired and checked by reading, not compiled, not run on hardware
//
#include <hip/hip_runtime.h>
#include <hip/hip_bf16.h>
#include <math.h>

// ---------------------------------------------------------------------------
// Problem dims (from reference): B=512, T=256, D=128, H=512, L=4, LBL=16
// ---------------------------------------------------------------------------
#define B_   512
#define T_   256
#define D_   128
#define H_   512
#define G4H  2048      // 4*H
#define L_   4
#define LBL_ 16

#define LDSPAD 40      // padded row stride (halves) for 32-wide K chunks: 80B rows
                       // -> fragment loads start on every 4th of 64 LDS banks
#define TILE_ELEMS (128 * LDSPAD)

typedef __attribute__((ext_vector_type(16))) _Float16 v16h;
typedef __attribute__((ext_vector_type(8)))  _Float16 v8h;
typedef __attribute__((ext_vector_type(8)))  float    v8f;

// ---------------------------------------------------------------------------
// WMMA fragment loaders (CDNA5 ISA 7.12.2 layouts, f16, wave32)
//
// A (16x32, MxK), row-major source with row stride ld (halves):
//   lanes 0-15 : row M=lane,   v0..v3 hold K=0..7,  v4..v7 hold K=16..23
//   lanes16-31 : row M=lane-16,v0..v3 hold K=8..15, v4..v7 hold K=24..31
// B (32x16, KxN) loaded from row-major W[n][k] (i.e. B^T) with row stride ld:
//   lanes 0-15 : col N=lane,    v0..v3 hold K=0..7,  v4..v7 hold K=16..23
//   lanes16-31 : col N=lane-16, v0..v3 hold K=8..15, v4..v7 hold K=24..31
// C/D (16x16 f32): VGPR r: lanes 0-15 -> (M=r, N=lane); lanes 16-31 -> (M=r+8)
// ---------------------------------------------------------------------------
__device__ __forceinline__ v16h load_fragA(const _Float16* tile, int ld) {
  const int lane = threadIdx.x & 31;
  const _Float16* p = tile + (lane & 15) * ld + ((lane >> 4) << 3);
  v8h lo = *(const v8h*)(p);        // K = kh+0..7
  v8h hi = *(const v8h*)(p + 16);   // K = kh+16..23
  v16h f;
#pragma unroll
  for (int j = 0; j < 8; ++j) { f[j] = lo[j]; f[8 + j] = hi[j]; }
  return f;
}

__device__ __forceinline__ v16h load_fragB(const _Float16* tile, int ld) {
  const int lane = threadIdx.x & 31;
  const _Float16* p = tile + (lane & 15) * ld + ((lane >> 4) << 3);
  v8h lo = *(const v8h*)(p);        // K = kh+0..7
  v8h hi = *(const v8h*)(p + 16);   // K = kh+16..23
  v16h f;
#pragma unroll
  for (int j = 0; j < 8; ++j) { f[j] = lo[j]; f[8 + j] = hi[j]; }
  return f;
}

__device__ __forceinline__ v8f wmma_f16(v16h a, v16h b, v8f c) {
  // (neg_a, A, neg_b, B, c_mod, C, reuse_a, reuse_b)
  return __builtin_amdgcn_wmma_f32_16x16x32_f16(false, a, false, b, (short)0, c,
                                                false, false);
}

__device__ __forceinline__ float sigmoidf_(float x) {
  return 1.0f / (1.0f + __expf(-x));
}

// --- double-buffered staging helpers: 128 rows x 32 halves per tile ---------
// Each of the 256 threads handles 2 x 16B vectors (row = vec>>2, col = (vec&3)*8).
__device__ __forceinline__ void chunk_load(const _Float16* srcA, int lda,
                                           const _Float16* srcB, int ldb,
                                           v8h* ra, v8h* rb) {
  const int tid = threadIdx.x;
#pragma unroll
  for (int it = 0; it < 2; ++it) {
    const int vec = tid + it * 256;
    const int r  = vec >> 2;
    const int cv = (vec & 3) << 3;
    ra[it] = *(const v8h*)(srcA + (size_t)r * lda + cv);
    rb[it] = *(const v8h*)(srcB + (size_t)r * ldb + cv);
  }
}

__device__ __forceinline__ void chunk_store(_Float16* Ad, _Float16* Bd,
                                            const v8h* ra, const v8h* rb) {
  const int tid = threadIdx.x;
#pragma unroll
  for (int it = 0; it < 2; ++it) {
    const int vec = tid + it * 256;
    const int r  = vec >> 2;
    const int cv = (vec & 3) << 3;
    *(v8h*)(Ad + r * LDSPAD + cv) = ra[it];
    *(v8h*)(Bd + r * LDSPAD + cv) = rb[it];
  }
}

// ---------------------------------------------------------------------------
// Converts / small setup kernels
// ---------------------------------------------------------------------------
__global__ void f32_to_f16_k(const float* __restrict__ s, _Float16* __restrict__ d, int n) {
  int i = blockIdx.x * blockDim.x + threadIdx.x;
  if (i < n) d[i] = (_Float16)s[i];
}

__global__ void bias_sum_k(const float* __restrict__ bih, const float* __restrict__ bhh,
                           float* __restrict__ bias, int n) {
  int i = blockIdx.x * blockDim.x + threadIdx.x;
  if (i < n) bias[i] = bih[i] + bhh[i];
}

__global__ void init_state_k(float* __restrict__ C, _Float16* __restrict__ Hs) {
  int i = blockIdx.x * blockDim.x + threadIdx.x;   // B*H threads
  C[i] = 0.0f;
  Hs[i] = (_Float16)0.0f;
}

// ---------------------------------------------------------------------------
// Fused LSTM step GEMM (LDS double-buffered, software pipelined):
//   G[b, 0..4H) = x_t @ Wih.T + h @ Whh.T + bias
//   M = B (512), N = 4H (2048), K = inK (phase 1) + H (phase 2)
// Workgroup = 256 threads = 8 waves; WG tile 128x128 (waves 4x2, each 32x64).
// Per 32-wide K chunk: prefetch chunk c+1 (global->regs) before the barrier,
// compute chunk c from LDS buf[c&1] (2 A-frags x 4 B-frags = 8 WMMAs/wave),
// then ds_store the prefetched regs into buf[(c+1)&1].
// Grid: 64 blocks (4 m-blocks x 16 n-blocks).
// ---------------------------------------------------------------------------
__global__ void __launch_bounds__(256)
lstm_step_gemm(const _Float16* __restrict__ X, int inK,
               const _Float16* __restrict__ Hs,
               const _Float16* __restrict__ Wih,
               const _Float16* __restrict__ Whh,
               const float* __restrict__ bias,
               float* __restrict__ G, int t) {
  __shared__ _Float16 Ash[2 * TILE_ELEMS];   // 2 x 10 KB
  __shared__ _Float16 Bsh[2 * TILE_ELEMS];   // 2 x 10 KB

  const int tid = threadIdx.x;
  const int wv = tid >> 5;        // 0..7
  const int wM = wv & 3;          // 4 m-slices of 32
  const int wN = wv >> 2;         // 2 n-slices of 64
  const int bM = blockIdx.x & 3;  // 4 blocks cover M=512
  const int bN = blockIdx.x >> 2; // 16 blocks cover N=2048
  const int m0 = bM * 128;
  const int n0 = bN * 128;

  v8f acc[2][4] = {{{}, {}, {}, {}}, {{}, {}, {}, {}}};

  // phase 1 sources: x_t @ Wih.T     phase 2 sources: h_{t-1} @ Whh.T
  const int ldx = T_ * inK;  // row stride between batch rows of X[:, t, :]
  const _Float16* Xbase = X + (size_t)m0 * ldx + (size_t)t * inK;
  const _Float16* Wibase = Wih + (size_t)n0 * inK;
  const _Float16* Hbase = Hs + (size_t)m0 * H_;
  const _Float16* Whbase = Whh + (size_t)n0 * H_;

  const int nck1 = inK / 32;
  const int nck = nck1 + H_ / 32;

  // prologue: stage chunk 0 into buffer 0
  v8h ra[2], rb[2];
  chunk_load(Xbase, ldx, Wibase, inK, ra, rb);
  chunk_store(Ash, Bsh, ra, rb);

  int cur = 0;
  for (int c = 0; c < nck; ++c) {
    const int nxt = c + 1;
    // prefetch next chunk into registers (global loads overlap the barrier +
    // this iteration's WMMAs)
    if (nxt < nck) {
      if (nxt < nck1) {
        chunk_load(Xbase + nxt * 32, ldx, Wibase + nxt * 32, inK, ra, rb);
      } else {
        const int k0 = (nxt - nck1) * 32;
        chunk_load(Hbase + k0, H_, Whbase + k0, H_, ra, rb);
      }
    }

    __syncthreads();  // buf[cur] stores (from prev iteration) visible

    const _Float16* Ac = Ash + cur * TILE_ELEMS;
    const _Float16* Bc = Bsh + cur * TILE_ELEMS;
    v16h a0 = load_fragA(Ac + (wM * 32) * LDSPAD, LDSPAD);
    v16h a1 = load_fragA(Ac + (wM * 32 + 16) * LDSPAD, LDSPAD);
    v16h b0 = load_fragB(Bc + (wN * 64) * LDSPAD, LDSPAD);
    v16h b1 = load_fragB(Bc + (wN * 64 + 16) * LDSPAD, LDSPAD);
    v16h b2 = load_fragB(Bc + (wN * 64 + 32) * LDSPAD, LDSPAD);
    v16h b3 = load_fragB(Bc + (wN * 64 + 48) * LDSPAD, LDSPAD);
    acc[0][0] = wmma_f16(a0, b0, acc[0][0]);
    acc[0][1] = wmma_f16(a0, b1, acc[0][1]);
    acc[0][2] = wmma_f16(a0, b2, acc[0][2]);
    acc[0][3] = wmma_f16(a0, b3, acc[0][3]);
    acc[1][0] = wmma_f16(a1, b0, acc[1][0]);
    acc[1][1] = wmma_f16(a1, b1, acc[1][1]);
    acc[1][2] = wmma_f16(a1, b2, acc[1][2]);
    acc[1][3] = wmma_f16(a1, b3, acc[1][3]);

    // store prefetched chunk into the other buffer (safe: all waves' reads of
    // that buffer finished before they signaled the barrier above; per-wave
    // LDS ops are in-order so these stores cannot pass this wave's loads)
    if (nxt < nck) {
      const int nb = cur ^ 1;
      chunk_store(Ash + nb * TILE_ELEMS, Bsh + nb * TILE_ELEMS, ra, rb);
      cur = nb;
    }
  }

  // ---- epilogue: + bias, store fp32 gate pre-activations ----
  const int lane = tid & 31;
  const int n = lane & 15;
  const int mb = (lane >> 4) << 3;
#pragma unroll
  for (int j = 0; j < 4; ++j) {
    const int nc = n0 + wN * 64 + j * 16 + n;
    const float bn = bias[nc];
#pragma unroll
    for (int i = 0; i < 2; ++i) {
      const int rbase = m0 + wM * 32 + i * 16 + mb;
#pragma unroll
      for (int r = 0; r < 8; ++r) {
        G[(size_t)(rbase + r) * G4H + nc] = acc[i][j][r] + bn;
      }
    }
  }
}

// ---------------------------------------------------------------------------
// LSTM cell elementwise update (PyTorch gate order i,f,g,o)
// ---------------------------------------------------------------------------
__global__ void __launch_bounds__(256)
lstm_cell(const float* __restrict__ G, float* __restrict__ C,
          _Float16* __restrict__ Hs, _Float16* __restrict__ Y, int t) {
  const int idx = blockIdx.x * blockDim.x + threadIdx.x;  // B*H threads
  const int b = idx >> 9;          // /H
  const int j = idx & (H_ - 1);
  const float* g = G + (size_t)b * G4H;
  const float gi = sigmoidf_(g[j]);
  const float gf = sigmoidf_(g[H_ + j]);
  const float gg = tanhf(g[2 * H_ + j]);
  const float go = sigmoidf_(g[3 * H_ + j]);
  const float c = gf * C[idx] + gi * gg;
  C[idx] = c;
  const float h = go * tanhf(c);
  Hs[idx] = (_Float16)h;
  Y[((size_t)b * T_ + t) * H_ + j] = (_Float16)h;
}

// ---------------------------------------------------------------------------
// Generic f16 WMMA GEMM: D = act(A @ W.T + bias), A:[M,K] ld=lda, W:[N,K] ld=ldb
// act==1 -> leaky_relu(0.2). Wave computes 32x32 tile; used only for the small
// fc layer (512x512x640), so the simple per-wave version is fine.
// ---------------------------------------------------------------------------
__global__ void __launch_bounds__(256)
gemm_f16_act(const _Float16* __restrict__ A, int lda,
             const _Float16* __restrict__ W, int ldb,
             const float* __restrict__ bias,
             float* __restrict__ D, int ldd,
             int K, int nTiles, int act) {
  const int wave = (blockIdx.x * blockDim.x + threadIdx.x) >> 5;
  const int mT = wave / nTiles;
  const int nT = wave % nTiles;
  const int m0 = mT * 32, n0 = nT * 32;

  v8f c00 = {}, c01 = {}, c10 = {}, c11 = {};
  const _Float16* A0 = A + (size_t)m0 * lda;
  const _Float16* A1 = A + (size_t)(m0 + 16) * lda;
  const _Float16* B0 = W + (size_t)n0 * ldb;
  const _Float16* B1 = W + (size_t)(n0 + 16) * ldb;
  for (int k0 = 0; k0 < K; k0 += 32) {
    v16h a0 = load_fragA(A0 + k0, lda);
    v16h a1 = load_fragA(A1 + k0, lda);
    v16h b0 = load_fragB(B0 + k0, ldb);
    v16h b1 = load_fragB(B1 + k0, ldb);
    c00 = wmma_f16(a0, b0, c00);
    c01 = wmma_f16(a0, b1, c01);
    c10 = wmma_f16(a1, b0, c10);
    c11 = wmma_f16(a1, b1, c11);
  }
  const int lane = threadIdx.x & 31;
  const int n = lane & 15;
  const int mb = (lane >> 4) << 3;
  const float bn0 = bias[n0 + n];
  const float bn1 = bias[n0 + 16 + n];
#pragma unroll
  for (int r = 0; r < 8; ++r) {
    float v00 = c00[r] + bn0, v01 = c01[r] + bn1;
    float v10 = c10[r] + bn0, v11 = c11[r] + bn1;
    if (act == 1) {
      v00 = v00 > 0.f ? v00 : 0.2f * v00;
      v01 = v01 > 0.f ? v01 : 0.2f * v01;
      v10 = v10 > 0.f ? v10 : 0.2f * v10;
      v11 = v11 > 0.f ? v11 : 0.2f * v11;
    }
    D[(size_t)(m0 + mb + r) * ldd + n0 + n]           = v00;
    D[(size_t)(m0 + mb + r) * ldd + n0 + 16 + n]      = v01;
    D[(size_t)(m0 + 16 + mb + r) * ldd + n0 + n]      = v10;
    D[(size_t)(m0 + 16 + mb + r) * ldd + n0 + 16 + n] = v11;
  }
}

// ---------------------------------------------------------------------------
// Length branch: one_hot(length-1) @ W_len.T == gather column; + b_len, ReLU.
// Writes f16 into combined[:, 0:128].
// ---------------------------------------------------------------------------
__global__ void length_branch_k(const int* __restrict__ length,
                                const float* __restrict__ W_len,
                                const float* __restrict__ b_len,
                                _Float16* __restrict__ comb) {
  const int idx = blockIdx.x * blockDim.x + threadIdx.x;  // B*128
  const int b = idx >> 7;
  const int j = idx & 127;
  const int li = length[b] - 1;
  float v = W_len[(size_t)j * T_ + li] + b_len[j];
  comb[(size_t)b * (128 + H_) + j] = (_Float16)(v > 0.f ? v : 0.f);
}

// Gather last hidden state at t = length-1 into combined[:, 128:640].
__global__ void gather_last_k(const _Float16* __restrict__ Y,
                              const int* __restrict__ length,
                              _Float16* __restrict__ comb) {
  const int idx = blockIdx.x * blockDim.x + threadIdx.x;  // B*H
  const int b = idx >> 9;
  const int j = idx & (H_ - 1);
  const int li = length[b] - 1;
  comb[(size_t)b * (128 + H_) + 128 + j] = Y[((size_t)b * T_ + li) * H_ + j];
}

// ---------------------------------------------------------------------------
// Label-routed head: idx=argmax(label[b]); out[b] = dot(hidden[b], W_heads[idx]) + b_heads[idx]
// One wave per batch row.
// ---------------------------------------------------------------------------
__global__ void __launch_bounds__(256)
head_k(const float* __restrict__ hidden, const float* __restrict__ label,
       const float* __restrict__ Wh, const float* __restrict__ bh,
       float* __restrict__ out) {
  const int wave = (blockIdx.x * blockDim.x + threadIdx.x) >> 5;
  const int lane = threadIdx.x & 31;
  const int b = wave;  // grid sized exactly: 64 blocks * 8 waves = 512
  const float* lb = label + (size_t)b * LBL_;
  int idx = 0;
  float best = lb[0];
#pragma unroll
  for (int j = 1; j < LBL_; ++j) {
    float v = lb[j];
    if (v > best) { best = v; idx = j; }
  }
  const float* w = Wh + (size_t)idx * H_;
  const float* h = hidden + (size_t)b * H_;
  float s = 0.f;
  for (int j = lane; j < H_; j += 32) s += h[j] * w[j];
#pragma unroll
  for (int off = 16; off > 0; off >>= 1) s += __shfl_down(s, off, 32);
  if (lane == 0) out[b] = s + bh[idx];
}

// ---------------------------------------------------------------------------
// Host launcher
// ---------------------------------------------------------------------------
extern "C" void kernel_launch(void* const* d_in, const int* in_sizes, int n_in,
                              void* d_out, int out_size, void* d_ws, size_t ws_size,
                              hipStream_t stream) {
  (void)in_sizes; (void)n_in; (void)out_size; (void)ws_size;

  const float* label    = (const float*)d_in[0];   // [B,16]
  const float* seq      = (const float*)d_in[1];   // [B,T,D]
  const int*   length   = (const int*)  d_in[2];   // [B]
  const float* Wih0     = (const float*)d_in[3];   // [4H,D]
  const float* Wih_rest = (const float*)d_in[4];   // [3,4H,H]
  const float* Whh      = (const float*)d_in[5];   // [4,4H,H]
  const float* bih      = (const float*)d_in[6];   // [4,4H]
  const float* bhh      = (const float*)d_in[7];   // [4,4H]
  const float* W_len    = (const float*)d_in[8];   // [128,T]
  const float* b_len    = (const float*)d_in[9];   // [128]
  const float* W_fc     = (const float*)d_in[10];  // [512,640]
  const float* b_fc     = (const float*)d_in[11];  // [512]
  const float* W_heads  = (const float*)d_in[12];  // [16,512]
  const float* b_heads  = (const float*)d_in[13];  // [16]
  float* out = (float*)d_out;                      // [B,1]

  // --- workspace bump allocator (256B aligned) ---
  char* ws = (char*)d_ws;
  size_t off = 0;
  auto alloc = [&](size_t bytes) -> char* {
    char* p = ws + off;
    off = (off + bytes + 255) & ~(size_t)255;
    return p;
  };
  _Float16* seq16   = (_Float16*)alloc((size_t)B_ * T_ * D_ * 2);        // 32 MiB
  _Float16* wih0_16 = (_Float16*)alloc((size_t)G4H * D_ * 2);
  _Float16* wihr_16 = (_Float16*)alloc((size_t)3 * G4H * H_ * 2);
  _Float16* whh16   = (_Float16*)alloc((size_t)L_ * G4H * H_ * 2);
  _Float16* wfc16   = (_Float16*)alloc((size_t)512 * 640 * 2);
  float*    bias    = (float*)   alloc((size_t)L_ * G4H * 4);
  _Float16* Y0      = (_Float16*)alloc((size_t)B_ * T_ * H_ * 2);        // 128 MiB
  _Float16* Y1      = (_Float16*)alloc((size_t)B_ * T_ * H_ * 2);        // 128 MiB
  float*    G       = (float*)   alloc((size_t)B_ * G4H * 4);            // 4 MiB
  float*    Cst     = (float*)   alloc((size_t)B_ * H_ * 4);
  _Float16* Hst     = (_Float16*)alloc((size_t)B_ * H_ * 2);
  _Float16* comb16  = (_Float16*)alloc((size_t)B_ * (128 + H_) * 2);
  float*    hidden  = (float*)   alloc((size_t)B_ * 512 * 4);

  auto cvt = [&](const float* s, _Float16* d, int n) {
    f32_to_f16_k<<<(n + 255) / 256, 256, 0, stream>>>(s, d, n);
  };

  // --- one-time conversions (re-done every call; deterministic) ---
  cvt(seq,      seq16,   B_ * T_ * D_);
  cvt(Wih0,     wih0_16, G4H * D_);
  cvt(Wih_rest, wihr_16, 3 * G4H * H_);
  cvt(Whh,      whh16,   L_ * G4H * H_);
  cvt(W_fc,     wfc16,   512 * 640);
  bias_sum_k<<<(L_ * G4H + 255) / 256, 256, 0, stream>>>(bih, bhh, bias, L_ * G4H);

  // --- 4-layer LSTM ---
  const _Float16* Xl = seq16;
  int inK = D_;
  _Float16* Ybufs[2] = {Y0, Y1};
  for (int l = 0; l < L_; ++l) {
    const _Float16* wih = (l == 0) ? wih0_16 : wihr_16 + (size_t)(l - 1) * G4H * H_;
    const _Float16* whh = whh16 + (size_t)l * G4H * H_;
    const float* bl = bias + (size_t)l * G4H;
    _Float16* Yl = Ybufs[l & 1];

    init_state_k<<<(B_ * H_) / 256, 256, 0, stream>>>(Cst, Hst);
    for (int t = 0; t < T_; ++t) {
      // 64 WGs x 8 waves: 128x128 double-buffered LDS tiles
      lstm_step_gemm<<<64, 256, 0, stream>>>(Xl, inK, Hst, wih, whh, bl, G, t);
      lstm_cell<<<(B_ * H_) / 256, 256, 0, stream>>>(G, Cst, Hst, Yl, t);
    }
    Xl = Yl;
    inK = H_;
  }

  // --- heads ---
  length_branch_k<<<(B_ * 128) / 256, 256, 0, stream>>>(length, W_len, b_len, comb16);
  gather_last_k<<<(B_ * H_) / 256, 256, 0, stream>>>(Xl, length, comb16);

  // hidden = leaky_relu(combined @ W_fc.T + b_fc): M=512,N=512,K=640
  // waves = (512/32)*(512/32) = 256 -> 32 blocks of 8 waves; nTiles = 16
  gemm_f16_act<<<32, 256, 0, stream>>>(comb16, 128 + H_, wfc16, 128 + H_,
                                       b_fc, hidden, 512, 128 + H_, 16, 1);

  // validity: one wave per batch row
  head_k<<<B_ / 8, 256, 0, stream>>>(hidden, label, W_heads, b_heads, out);
}
